// GatedLinearAttention_21878563405930
// MI455X (gfx1250) — matrix operand order repaired and checked by reading, not compiled, not run on hardware
//
#include <hip/hip_runtime.h>
#include <hip/hip_bf16.h>

typedef __attribute__((ext_vector_type(16))) _Float16 v16h;
typedef __attribute__((ext_vector_type(8)))  float    v8f;

#define Bz   4
#define Tz   1024
#define Dz   1024
#define Hz   8
#define KDz  512
#define VDz  1024
#define DKz  64
#define DVz  128
#define Mz   (Bz * Tz)          // 4096 rows
#define CONVz 4

// ---------------------------------------------------------------------------
// Conversions
// ---------------------------------------------------------------------------
__global__ void cvt_f16_kernel(const float* __restrict__ in,
                               _Float16* __restrict__ out, size_t n) {
    size_t i = (size_t)blockIdx.x * blockDim.x + threadIdx.x;
    if (i < n) out[i] = (_Float16)in[i];
}

// in: K x N row-major  ->  out: N x K row-major (transposed), f16
__global__ void cvt_f16_tr_kernel(const float* __restrict__ in,
                                  _Float16* __restrict__ out, int K, int N) {
    size_t i = (size_t)blockIdx.x * blockDim.x + threadIdx.x;
    if (i >= (size_t)K * N) return;
    int k = (int)(i / N);
    int n = (int)(i % N);
    out[(size_t)n * K + k] = (_Float16)in[i];
}

// ---------------------------------------------------------------------------
// WMMA GEMM: C(MxN,f32) = A(MxK,f16 row-major) * B(KxN), B supplied as
// Bt = B^T (N x K, f16 row-major).
// Block = 4 waves arranged 2x2; wave tile = 32x64 (2x4 WMMA accumulators);
// block tile = 64x128. N, K compile-time => immediate-offset addressing.
// All dims assumed multiples of the tile (true for this problem).
// ---------------------------------------------------------------------------
template <int N, int K>
__global__ void __launch_bounds__(128)
gemm_wmma_kernel(const _Float16* __restrict__ A,
                 const _Float16* __restrict__ Bt,
                 float* __restrict__ C) {
    const int wave = threadIdx.x >> 5;
    const int lane = threadIdx.x & 31;
    const int lr   = lane & 15;          // row (A) / col (B) within 16
    const int hi   = lane >> 4;          // K-half select

    const int bm = blockIdx.y * 64  + (wave >> 1) * 32;
    const int bn = blockIdx.x * 128 + (wave & 1) * 64;

    v8f acc[2][4];
    #pragma unroll
    for (int i = 0; i < 2; ++i)
        #pragma unroll
        for (int j = 0; j < 4; ++j)
            acc[i][j] = (v8f){0.f,0.f,0.f,0.f,0.f,0.f,0.f,0.f};

    const _Float16* ap0 = A  + (size_t)(bm + lr) * K + hi * 8;
    const _Float16* bp0 = Bt + (size_t)(bn + lr) * K + hi * 8;

    for (int k0 = 0; k0 < K; k0 += 32) {
        v16h fa[2], fb[4];
        // A fragments: lane<16 -> K = k0+{0..7,16..23}; lane>=16 -> +8
        #pragma unroll
        for (int i = 0; i < 2; ++i) {
            const _Float16* ap = ap0 + (size_t)(i * 16) * K + k0;
            #pragma unroll
            for (int e = 0; e < 8; ++e) fa[i][e]     = ap[e];
            #pragma unroll
            for (int e = 0; e < 8; ++e) fa[i][8 + e] = ap[16 + e];
        }
        // B fragments: lane<16 -> col lr, K = k0+{0..7,16..23}; lane>=16 -> +8
        #pragma unroll
        for (int j = 0; j < 4; ++j) {
            const _Float16* bp = bp0 + (size_t)(j * 16) * K + k0;
            #pragma unroll
            for (int e = 0; e < 8; ++e) fb[j][e]     = bp[e];
            #pragma unroll
            for (int e = 0; e < 8; ++e) fb[j][8 + e] = bp[16 + e];
        }
        #pragma unroll
        for (int i = 0; i < 2; ++i)
            #pragma unroll
            for (int j = 0; j < 4; ++j)
                acc[i][j] = __builtin_amdgcn_wmma_f32_16x16x32_f16(
                    false, fa[i], false, fb[j], (short)0, acc[i][j], false, false);
    }

    // C layout: element e -> row (hi*8 + e), col lr within each 16x16 tile
    #pragma unroll
    for (int i = 0; i < 2; ++i)
        #pragma unroll
        for (int j = 0; j < 4; ++j) {
            float* cp = C + (size_t)(bm + i * 16 + hi * 8) * N
                          + (bn + j * 16 + lr);
            #pragma unroll
            for (int e = 0; e < 8; ++e)
                cp[(size_t)e * N] = acc[i][j][e];
        }
}

// ---------------------------------------------------------------------------
// Depthwise causal conv1d (K=4, cross-correlation) + SiLU, optional token mask
// x: (B,T,C) f32 -> y: (B,T,C) f32
// ---------------------------------------------------------------------------
__global__ void conv_silu_kernel(const float* __restrict__ x,
                                 const float* __restrict__ w,   // (C, 4)
                                 float* __restrict__ y,
                                 int C, int applyMask,
                                 const unsigned char* __restrict__ mask) {
    size_t i = (size_t)blockIdx.x * blockDim.x + threadIdx.x;
    if (i >= (size_t)Mz * C) return;
    int c = (int)(i % C);
    size_t bt = i / C;
    int t = (int)(bt % Tz);
    float acc = 0.f;
    #pragma unroll
    for (int j = 0; j < CONVz; ++j) {
        int ts = t - (CONVz - 1) + j;
        if (ts >= 0)
            acc += x[(bt - (size_t)(CONVz - 1 - j)) * C + c] * w[c * CONVz + j];
    }
    float s = acc / (1.f + __expf(-acc));   // SiLU
    if (applyMask && mask[bt] == 0) s = 0.f;
    y[i] = s;
}

// ---------------------------------------------------------------------------
// Low-rank gate:  glr = src @ Wg1   (M x 16)
// ---------------------------------------------------------------------------
__global__ void lowrank_proj_kernel(const float* __restrict__ src,
                                    const float* __restrict__ Wg1,
                                    float* __restrict__ glr) {
    size_t i = (size_t)blockIdx.x * blockDim.x + threadIdx.x;
    if (i >= (size_t)Mz * 16) return;
    int r = (int)(i & 15);
    size_t m = i >> 4;
    const float* s = src + m * Dz;
    float acc = 0.f;
    for (int d = 0; d < Dz; ++d) acc += s[d] * Wg1[(size_t)d * 16 + r];
    glr[i] = acc;
}

// gk = log_sigmoid(glr @ Wg2 + bg2) / 16    (M x KD)
__global__ void gate_kernel(const float* __restrict__ glr,
                            const float* __restrict__ Wg2,
                            const float* __restrict__ bg2,
                            float* __restrict__ gko) {
    size_t i = (size_t)blockIdx.x * blockDim.x + threadIdx.x;
    if (i >= (size_t)Mz * KDz) return;
    int n = (int)(i % KDz);
    size_t m = i / KDz;
    float s = bg2[n];
    #pragma unroll
    for (int r = 0; r < 16; ++r)
        s += glr[m * 16 + r] * Wg2[(size_t)r * KDz + n];
    float ls = (s >= 0.f) ? -log1pf(__expf(-s)) : (s - log1pf(__expf(s)));
    gko[i] = ls * (1.f / 16.f);
}

// ---------------------------------------------------------------------------
// GLA recurrent scan. One block per (b,h). 256 threads.
// Thread t owns v-column (t & 127) and k-slice [(t>>7)*32, +32) of S (regs).
// ---------------------------------------------------------------------------
__global__ void __launch_bounds__(256)
gla_scan_kernel(const float* __restrict__ qc, const float* __restrict__ kc,
                const float* __restrict__ vc, const float* __restrict__ gk,
                float* __restrict__ o) {
    const int bh = blockIdx.x;
    const int b = bh / Hz, h = bh % Hz;
    const int tid = threadIdx.x;
    const int v = tid & (DVz - 1);
    const int kbase = (tid >> 7) * 32;

    __shared__ float sQ[DKz], sK[DKz], sG[DKz], sV[DVz], sPart[256];

    float S[32];
    #pragma unroll
    for (int i = 0; i < 32; ++i) S[i] = 0.f;

    const size_t baseQK = (size_t)b * Tz * KDz + (size_t)h * DKz;
    const size_t baseV  = (size_t)b * Tz * VDz + (size_t)h * DVz;

    for (int t = 0; t < Tz; ++t) {
        if (tid < DKz) {
            size_t idx = baseQK + (size_t)t * KDz + tid;
            sK[tid] = kc[idx];
            sQ[tid] = qc[idx] * 0.125f;          // DK^-0.5 = 1/8
            sG[tid] = __expf(gk[idx]);
        } else if (tid < DKz + DVz) {
            sV[tid - DKz] = vc[baseV + (size_t)t * VDz + (tid - DKz)];
        }
        __syncthreads();

        const float vv = sV[v];
        float part = 0.f;
        #pragma unroll
        for (int i = 0; i < 32; ++i) {
            const int k = kbase + i;
            S[i] = S[i] * sG[k] + sK[k] * vv;
            part += sQ[k] * S[i];
        }
        sPart[tid] = part;
        __syncthreads();

        if (tid < DVz)
            o[baseV + (size_t)t * VDz + v] = part + sPart[tid + DVz];
        // next-iteration loads touch only sQ/sK/sG/sV; sPart reads complete
        // before any thread can pass the first barrier of the next iteration.
    }
}

// ---------------------------------------------------------------------------
// Fused RMSNorm (over DV) + swish output gate, emit f16 for final GEMM.
// One block per (m, h); 128 threads = DV lanes.
// ---------------------------------------------------------------------------
__global__ void __launch_bounds__(128)
norm_gate_kernel(const float* __restrict__ o, const float* __restrict__ gproj,
                 const float* __restrict__ rms_w, _Float16* __restrict__ oh) {
    const int mh = blockIdx.x;
    const size_t m = (size_t)(mh / Hz);
    const int h = mh % Hz;
    const int v = threadIdx.x;
    const size_t idx = m * VDz + (size_t)h * DVz + v;

    __shared__ float red[DVz];
    float x = o[idx];
    red[v] = x * x;
    __syncthreads();
    #pragma unroll
    for (int s = DVz / 2; s > 0; s >>= 1) {
        if (v < s) red[v] += red[v + s];
        __syncthreads();
    }
    float r = rsqrtf(red[0] / (float)DVz + 1e-5f);
    float g = gproj[idx];
    float sg = g / (1.f + __expf(-g));           // g * sigmoid(g)
    oh[idx] = (_Float16)(x * r * rms_w[v] * sg);
}

// ---------------------------------------------------------------------------
// Host launcher
// ---------------------------------------------------------------------------
extern "C" void kernel_launch(void* const* d_in, const int* in_sizes, int n_in,
                              void* d_out, int out_size, void* d_ws, size_t ws_size,
                              hipStream_t stream) {
    (void)in_sizes; (void)n_in; (void)out_size; (void)ws_size;
    const float*         src     = (const float*)d_in[0];
    const unsigned char* vmask   = (const unsigned char*)d_in[1];
    const float*         Wq      = (const float*)d_in[2];
    const float*         Wk      = (const float*)d_in[3];
    const float*         Wv      = (const float*)d_in[4];
    const float*         conv_q  = (const float*)d_in[5];
    const float*         conv_k  = (const float*)d_in[6];
    const float*         conv_v  = (const float*)d_in[7];
    const float*         Wg1     = (const float*)d_in[8];
    const float*         Wg2     = (const float*)d_in[9];
    const float*         bg2     = (const float*)d_in[10];
    const float*         Wgate   = (const float*)d_in[11];
    const float*         rms_w   = (const float*)d_in[12];
    const float*         Wo      = (const float*)d_in[13];
    float* out = (float*)d_out;

    char* p = (char*)d_ws;
    auto alloc = [&](size_t bytes) -> void* {
        void* r = (void*)p;
        p += (bytes + 255) & ~(size_t)255;
        return r;
    };
    _Float16* sH     = (_Float16*)alloc((size_t)Mz * Dz * 2);
    _Float16* WqT    = (_Float16*)alloc((size_t)KDz * Dz * 2);
    _Float16* WkT    = (_Float16*)alloc((size_t)KDz * Dz * 2);
    _Float16* WvT    = (_Float16*)alloc((size_t)VDz * Dz * 2);
    _Float16* WgateT = (_Float16*)alloc((size_t)VDz * Dz * 2);
    _Float16* WoT    = (_Float16*)alloc((size_t)Dz * VDz * 2);
    float*    qp     = (float*)alloc((size_t)Mz * KDz * 4);
    float*    kp     = (float*)alloc((size_t)Mz * KDz * 4);
    float*    vp     = (float*)alloc((size_t)Mz * VDz * 4);
    float*    qcb    = (float*)alloc((size_t)Mz * KDz * 4);
    float*    kcb    = (float*)alloc((size_t)Mz * KDz * 4);
    float*    vcb    = (float*)alloc((size_t)Mz * VDz * 4);
    float*    gproj  = (float*)alloc((size_t)Mz * VDz * 4);
    float*    glr    = (float*)alloc((size_t)Mz * 16 * 4);
    float*    gkb    = (float*)alloc((size_t)Mz * KDz * 4);
    float*    obuf   = (float*)alloc((size_t)Mz * VDz * 4);
    _Float16* oh     = (_Float16*)alloc((size_t)Mz * VDz * 2);

    const int TPB = 256;
    auto blocks = [&](size_t n) { return (unsigned)((n + TPB - 1) / TPB); };

    // 1) conversions
    cvt_f16_kernel<<<blocks((size_t)Mz * Dz), TPB, 0, stream>>>(src, sH, (size_t)Mz * Dz);
    cvt_f16_tr_kernel<<<blocks((size_t)Dz * KDz), TPB, 0, stream>>>(Wq, WqT, Dz, KDz);
    cvt_f16_tr_kernel<<<blocks((size_t)Dz * KDz), TPB, 0, stream>>>(Wk, WkT, Dz, KDz);
    cvt_f16_tr_kernel<<<blocks((size_t)Dz * VDz), TPB, 0, stream>>>(Wv, WvT, Dz, VDz);
    cvt_f16_tr_kernel<<<blocks((size_t)Dz * VDz), TPB, 0, stream>>>(Wgate, WgateT, Dz, VDz);
    cvt_f16_tr_kernel<<<blocks((size_t)VDz * Dz), TPB, 0, stream>>>(Wo, WoT, VDz, Dz);

    // 2) WMMA projections (block tile 64x128)
    dim3 gq(KDz / 128, Mz / 64), gv(VDz / 128, Mz / 64);
    gemm_wmma_kernel<KDz, Dz><<<gq, 128, 0, stream>>>(sH, WqT, qp);
    gemm_wmma_kernel<KDz, Dz><<<gq, 128, 0, stream>>>(sH, WkT, kp);
    gemm_wmma_kernel<VDz, Dz><<<gv, 128, 0, stream>>>(sH, WvT, vp);
    gemm_wmma_kernel<VDz, Dz><<<gv, 128, 0, stream>>>(sH, WgateT, gproj);

    // 3) short conv + SiLU (mask only on v)
    conv_silu_kernel<<<blocks((size_t)Mz * KDz), TPB, 0, stream>>>(qp, conv_q, qcb, KDz, 0, vmask);
    conv_silu_kernel<<<blocks((size_t)Mz * KDz), TPB, 0, stream>>>(kp, conv_k, kcb, KDz, 0, vmask);
    conv_silu_kernel<<<blocks((size_t)Mz * VDz), TPB, 0, stream>>>(vp, conv_v, vcb, VDz, 1, vmask);

    // 4) low-rank forget gate
    lowrank_proj_kernel<<<blocks((size_t)Mz * 16), TPB, 0, stream>>>(src, Wg1, glr);
    gate_kernel<<<blocks((size_t)Mz * KDz), TPB, 0, stream>>>(glr, Wg2, bg2, gkb);

    // 5) recurrent scan
    gla_scan_kernel<<<Bz * Hz, 256, 0, stream>>>(qcb, kcb, vcb, gkb, obuf);

    // 6) RMSNorm + swish gate -> f16
    norm_gate_kernel<<<Mz * Hz, DVz, 0, stream>>>(obuf, gproj, rms_w, oh);

    // 7) output projection (WMMA) -> d_out
    dim3 go(Dz / 128, Mz / 64);
    gemm_wmma_kernel<Dz, VDz><<<go, 128, 0, stream>>>(oh, WoT, out);
}
